// Morphology_180388626682
// MI455X (gfx1250) — matrix operand order repaired and checked by reading, not compiled, not run on hardware
//
#include <hip/hip_runtime.h>

// ---- CDNA5 (gfx1250) morphological dilation via exp->conv(WMMA bf16)->log ----
//
// out[b,o,h,w] = (1/B)*log( sum_{c,ki,kj} exp(B*(xpad + w)) )
//             = (1/B)*( log( conv2d( exp(B*x - SHIFT), exp(B*w) ) ) + SHIFT )
//
// GEMM per tile: C[M=16 (out-ch, 8 valid), N=16 pixels] += A[16xK'] * B[K'x16].
// K' re-enumerated in tap PAIRS so each B-fragment VGPR is one ds_load_b32:
//   per (c,ki) row: pairs (kj=0,1), (2,3), (4,pad)  ->  K' = 8*5*3*2 = 240 -> 256
//   = 8 chunks of 32 -> 8 x v_wmma_f32_16x16x32_bf16 per row-tile.
// Strip staging uses CDNA5 async copy (global_load_async_to_lds_b128, ASYNCcnt).

typedef __attribute__((ext_vector_type(16))) __bf16 v16bf;
typedef __attribute__((ext_vector_type(8)))  __bf16 v8bf;
typedef __attribute__((ext_vector_type(8)))  float  v8f;

#define BETA_F   15.0f
#define SHIFT_F  10.0f
#define LOG2E_F  1.4426950408889634f
#define LN2_F    0.6931471805599453f

#define HP   260      // padded height (256 + 2*2)
#define WPD  264      // padded width (256 + 2*2, rounded to 8 for b128 rows)
#define KP2  256      // padded pair-K: 240 valid -> 8 chunks of 32
#define NPAIR 128     // KP2/2 pair slots
#define SPITCH 136    // raw strip row pitch (bf16), 16B-aligned rows
#define SP2   136     // pair-strip row pitch (u32 entries)
#define ROWS 4        // output rows per block (amortize 5-row halo)

// ---------- kernel 1: weight exponentials, bf16, [16 o][256 k'] ----------
__global__ __launch_bounds__(256) void prep_w(const float* __restrict__ wgt,
                                              __bf16* __restrict__ Fw) {
    int idx = blockIdx.x * 256 + threadIdx.x;
    if (idx >= 16 * KP2) return;
    int o = idx / KP2;
    int k = idx - o * KP2;
    int p = k >> 1, s = k & 1;       // pair slot, element within pair
    float v = 0.0f;
    if (o < 8 && p < 120) {
        int t  = p / 3, j = p - t * 3;   // t = c*5+ki, j = pair within row
        int kj = 2 * j + s;
        if (kj <= 4) {
            int c = t / 5, ki = t - c * 5;
            v = exp2f(wgt[((o * 8 + c) * 5 + ki) * 5 + kj] * (BETA_F * LOG2E_F));
        }
    }
    Fw[idx] = (__bf16)v;
}

// ---------- kernel 2: E = exp(B*x - SHIFT) into padded bf16 [16][8][260][264] ----------
__global__ __launch_bounds__(256) void prep_e(const float* __restrict__ x,
                                              __bf16* __restrict__ Epad, int total) {
    int idx = blockIdx.x * 256 + threadIdx.x;
    if (idx >= total) return;
    int wp = idx % WPD;
    int t  = idx / WPD;
    int hp = t % HP;
    int bc = t / HP;
    int hx = hp - 2, wx = wp - 2;
    float v;
    if ((unsigned)hx < 256u && (unsigned)wx < 256u)
        v = exp2f(fmaf(x[(size_t)bc * 65536 + hx * 256 + wx],
                       BETA_F * LOG2E_F, -SHIFT_F * LOG2E_F));
    else
        v = exp2f(-SHIFT_F * LOG2E_F);   // zero-padded x -> exp(B*0 - SHIFT)
    Epad[idx] = (__bf16)v;
}

// ---------- kernel 3: implicit-GEMM conv with WMMA + log epilogue ----------
__global__ __launch_bounds__(256) void conv_lse(const __bf16* __restrict__ Fw,
                                                const __bf16* __restrict__ Epad,
                                                float* __restrict__ out) {
    __shared__ __bf16 raw[64 * SPITCH];        // staged E strip, ~17 KB
    __shared__ unsigned strip2[64 * SP2];      // overlapping pairs (E[c],E[c+1]), ~34 KB
    __shared__ unsigned short poff[NPAIR];     // pair slot -> strip2 offset

    const int b     = blockIdx.z;
    const int hbase = blockIdx.y * ROWS;
    const int wbase = blockIdx.x * 128;
    const int tid   = threadIdx.x;

    // Stage raw E strip (64 rows x 136 bf16) with CDNA5 async copies.
    for (int i = tid; i < 64 * 17; i += 256) {
        int t  = i / 17;            // 0..63 = c*8 + rr
        int j  = i - t * 17;        // vec8 column
        int c  = t >> 3;
        int rr = t & 7;
        const __bf16* src =
            Epad + ((size_t)(b * 8 + c) * HP + (hbase + rr)) * WPD + wbase + j * 8;
        unsigned lds_off = (unsigned)(size_t)(void*)&raw[t * SPITCH + j * 8];
        asm volatile("global_load_async_to_lds_b128 %0, %1, off"
                     :: "v"(lds_off), "v"(src) : "memory");
    }
    // pair slot p -> strip2 element offset; p>=120 pads keep 0 (A operand is 0
    // there, so gathered finite garbage contributes 0*finite = 0).
    if (tid < NPAIR) {
        int p = tid;
        unsigned short off = 0;
        if (p < 120) {
            int t = p / 3, j = p - t * 3;
            int c = t / 5, ki = t - c * 5;
            off = (unsigned short)((c * 8 + ki) * SP2 + 2 * j);
        }
        poff[p] = off;
    }
    asm volatile("s_wait_asynccnt 0x0" ::: "memory");
    __syncthreads();

    // Build overlapping-pair strip with aligned u32 reads; one shift/or per odd pair.
    // strip2[row][c] = (E[row][c], E[row][c+1]); cols 0..133 built (reads use <=131).
    {
        const unsigned* raw32 = (const unsigned*)raw;   // row pitch 68 u32
        for (int i2 = tid; i2 < 64 * 67; i2 += 256) {
            int row = i2 / 67, i = i2 - row * 67;
            unsigned a  = raw32[row * 68 + i];
            unsigned bb = raw32[row * 68 + i + 1];
            uint2 pr;
            pr.x = a;                             // (E[2i],   E[2i+1])
            pr.y = (a >> 16) | (bb << 16);        // (E[2i+1], E[2i+2])
            *(uint2*)&strip2[row * SP2 + 2 * i] = pr;
        }
    }
    __syncthreads();

    const int lane = tid & 31;
    const int wv   = tid >> 5;          // 8 waves, 16 pixels each
    const int col  = lane & 15;
    const int px   = (wv << 4) | col;   // pixel within 128-wide strip
    const int kbB  = (lane & 16);       // B lane-half term (0|16); >>2 = pair-slot base (0|4)
    const int kbA  = (lane & 16) >> 1;  // A-operand K base (0|8)
    const __bf16* Arow = Fw + col * KP2;   // A row M = out-channel = lane&15

    v8f acc[ROWS];
#pragma unroll
    for (int ry = 0; ry < ROWS; ++ry)
        acc[ry] = (v8f){0.f, 0.f, 0.f, 0.f, 0.f, 0.f, 0.f, 0.f};

#pragma unroll
    for (int ch = 0; ch < 8; ++ch) {
        // A fragment: element e -> K = kbA + (e<8 ? e : e+8); two aligned 16B runs.
        union { v16bf v; v8bf h[2]; } A;
        A.h[0] = *(const v8bf*)(Arow + ch * 32 + kbA);
        A.h[1] = *(const v8bf*)(Arow + ch * 32 + kbA + 16);

        // 8 pair-gather bases per chunk (poff reads are two 8B runs per lane, matching A's K order).
        int pb[8];
#pragma unroll
        for (int r = 0; r < 8; ++r)
            pb[r] = (int)poff[ch * 16 + (kbB >> 2) + (r < 4 ? r : r + 4)] + px;

#pragma unroll
        for (int ry = 0; ry < ROWS; ++ry) {
            union { v16bf v; unsigned u[8]; } B;
#pragma unroll
            for (int r = 0; r < 8; ++r)
                B.u[r] = strip2[pb[r] + ry * SP2];   // one b32 = one B VGPR
            acc[ry] = __builtin_amdgcn_wmma_f32_16x16x32_bf16(
                false, A.v, false, B.v, (short)0, acc[ry], false, false);
        }
    }

    // D layout: lanes 0-15 hold rows M=0..7 (the 8 valid out-channels).
    if (lane < 16) {
#pragma unroll
        for (int ry = 0; ry < ROWS; ++ry) {
            const int h = hbase + ry;
            const int w = wbase + px;
#pragma unroll
            for (int o = 0; o < 8; ++o) {
                float s = acc[ry][o];
                out[((size_t)(b * 8 + o) * 256 + h) * 256 + w] =
                    log2f(s) * (LN2_F / BETA_F) + (SHIFT_F / BETA_F);
            }
        }
    }
}

extern "C" void kernel_launch(void* const* d_in, const int* in_sizes, int n_in,
                              void* d_out, int out_size, void* d_ws, size_t ws_size,
                              hipStream_t stream) {
    const float* x = (const float*)d_in[0];   // (16,8,256,256) f32
    const float* w = (const float*)d_in[1];   // (8,8,5,5) f32
    float* out = (float*)d_out;               // (16,8,256,256) f32

    __bf16* Fw   = (__bf16*)d_ws;                          // 16*256*2 = 8 KB
    __bf16* Epad = (__bf16*)((char*)d_ws + 8192);          // 16*8*260*264*2 ~ 17.6 MB

    prep_w<<<(16 * KP2 + 255) / 256, 256, 0, stream>>>(w, Fw);

    int totalE = 16 * 8 * HP * WPD;
    prep_e<<<(totalE + 255) / 256, 256, 0, stream>>>(x, Epad, totalE);

    dim3 grid(2, 256 / ROWS, 16);   // (w-strip, h-block, batch)
    conv_lse<<<grid, 256, 0, stream>>>(Fw, Epad, out);
}
